// CustomMSDeformableAttentionPyTorch_41747082117470
// MI455X (gfx1250) — compile-verified, matrix operand order checked by reading, not run on hardware
//
#include <hip/hip_runtime.h>

// ---------------------------------------------------------------------------
// Static problem config (mirrors reference)
// ---------------------------------------------------------------------------
#define BS   8
#define NQ   2000
#define NV   21760          // 128*128 + 64*64 + 32*32 + 16*16
#define CDIM 256
#define NH   8
#define HD   32
#define NL   4
#define NP   4
#define NROW (BS * NQ)      // 16000 query rows
#define MV   (BS * NV)      // 174080 value rows

typedef __attribute__((ext_vector_type(16))) __bf16 v16bf;
typedef __attribute__((ext_vector_type(8)))  float  v8f;

// ---------------------------------------------------------------------------
// Weight pre-swizzle: convert W (K x N, fp32 row-major) into bf16 fragments in
// the exact CDNA5 WMMA B-operand layout (ISA 7.12.2, 16-bit 32x16 operand):
//   lane 0-15  -> col = n_tile*16 + lane,    K = {0..7, 16..23}  + 32*ks
//   lane 16-31 -> col = n_tile*16 + lane-16, K = {8..15, 24..31} + 32*ks
// Packed address: ((ks * (N/16) + nt) * 32 + lane) * 16 bf16 elements, so the
// GEMM hot loop reads a lane's whole fragment as two global_load_b128.
// Packed bf16 weights are 448KB total: one-time cost, L2-resident thereafter.
// ---------------------------------------------------------------------------
__global__ __launch_bounds__(256)
void pack_weight_bf16(const float* __restrict__ W, __bf16* __restrict__ P,
                      int K, int N) {
    const int total = (K >> 5) * (N >> 4) * 32;
    const int t = blockIdx.x * blockDim.x + threadIdx.x;
    if (t >= total) return;
    const int lane = t & 31;
    const int tile = t >> 5;
    const int ntiles = N >> 4;
    const int nt = tile % ntiles;
    const int ks = tile / ntiles;

    const int c   = nt * 16 + (lane & 15);
    const int k0  = ks * 32 + ((lane < 16) ? 0 : 8);
    __bf16* dst = P + ((size_t)t) * 16;
#pragma unroll
    for (int e = 0; e < 8; ++e) dst[e]     = (__bf16)W[(size_t)(k0 + e) * N + c];
#pragma unroll
    for (int e = 0; e < 8; ++e) dst[8 + e] = (__bf16)W[(size_t)(k0 + 16 + e) * N + c];
}

// ---------------------------------------------------------------------------
// Fragment loaders
// ---------------------------------------------------------------------------
static __device__ __forceinline__ v16bf load_fragA(const float* __restrict__ A,
                                                   int lda, int lane, int k0) {
    const int r   = lane & 15;
    const int klo = (lane < 16) ? 0 : 8;
    const float* p = A + (size_t)r * lda + k0 + klo;   // 32B aligned
    const float4 a0 = *(const float4*)(p);
    const float4 a1 = *(const float4*)(p + 4);
    const float4 a2 = *(const float4*)(p + 16);
    const float4 a3 = *(const float4*)(p + 20);
    v16bf f;
    f[0]  = (__bf16)a0.x; f[1]  = (__bf16)a0.y; f[2]  = (__bf16)a0.z; f[3]  = (__bf16)a0.w;
    f[4]  = (__bf16)a1.x; f[5]  = (__bf16)a1.y; f[6]  = (__bf16)a1.z; f[7]  = (__bf16)a1.w;
    f[8]  = (__bf16)a2.x; f[9]  = (__bf16)a2.y; f[10] = (__bf16)a2.z; f[11] = (__bf16)a2.w;
    f[12] = (__bf16)a3.x; f[13] = (__bf16)a3.y; f[14] = (__bf16)a3.z; f[15] = (__bf16)a3.w;
    return f;
}

static __device__ __forceinline__ v16bf load_fragB_packed(const __bf16* __restrict__ Bp,
                                                          int ntiles, int nt, int lane, int ks) {
    const __bf16* p = Bp + ((((size_t)ks * ntiles + nt) << 5) + lane) * 16;  // 32B aligned
    v16bf f;
    __builtin_memcpy(&f, p, sizeof(f));   // two b128 loads
    return f;
}

// ---------------------------------------------------------------------------
// GEMM + bias:  D[M,N] = A[M,K] @ Bpacked + bias[N]
// fp32 A streamed + cvt_pk to bf16; weights pre-swizzled bf16; f32 accumulate.
// Block = 256 threads = 8 waves, 4(M) x 2(N); block tile 64x64.
// TK templatized (=256 here) so the k-loop fully unrolls.
// ---------------------------------------------------------------------------
template <bool OUT_BF16, int TK>
__global__ __launch_bounds__(256)
void gemm_bias_wmma(const float* __restrict__ A, const __bf16* __restrict__ Bp,
                    const float* __restrict__ bias, void* __restrict__ Dv,
                    int M, int N) {
    const int lane = threadIdx.x & 31;
    const int wid  = threadIdx.x >> 5;          // 0..7
    const int wm   = wid >> 1;                  // 0..3
    const int wn   = wid & 1;                   // 0..1
    const int row0 = blockIdx.y * 64 + wm * 16;
    const int col0 = blockIdx.x * 64 + wn * 32;
    const int ntiles = N >> 4;
    const int nt0 = col0 >> 4;

    v8f acc0 = {};
    v8f acc1 = {};

    const float* Arow = A + (size_t)row0 * TK;
#pragma unroll
    for (int ks = 0; ks < TK / 32; ++ks) {
        v16bf a  = load_fragA(Arow, TK, lane, ks * 32);
        v16bf b0 = load_fragB_packed(Bp, ntiles, nt0,     lane, ks);
        v16bf b1 = load_fragB_packed(Bp, ntiles, nt0 + 1, lane, ks);
        acc0 = __builtin_amdgcn_wmma_f32_16x16x32_bf16(false, a, false, b0,
                                                       (short)0, acc0, false, false);
        acc1 = __builtin_amdgcn_wmma_f32_16x16x32_bf16(false, a, false, b1,
                                                       (short)0, acc1, false, false);
    }

    // C/D layout: lanes 0-15 -> N=lane, M=i ; lanes 16-31 -> N=lane-16, M=i+8
    const int n0 = col0 + (lane & 15);
    const int n1 = n0 + 16;
    const int mo = (lane < 16) ? 0 : 8;
    const float bia0 = bias[n0];
    const float bia1 = bias[n1];
#pragma unroll
    for (int i = 0; i < 8; ++i) {
        const int m = row0 + mo + i;
        const float d0 = acc0[i] + bia0;
        const float d1 = acc1[i] + bia1;
        if (OUT_BF16) {
            __bf16* D = (__bf16*)Dv;
            D[(size_t)m * N + n0] = (__bf16)d0;
            D[(size_t)m * N + n1] = (__bf16)d1;
        } else {
            float* D = (float*)Dv;
            D[(size_t)m * N + n0] = d0;
            D[(size_t)m * N + n1] = d1;
        }
    }
}

// ---------------------------------------------------------------------------
// Softmax over groups of 16 (one (row, head) slice per thread).
// ---------------------------------------------------------------------------
__global__ __launch_bounds__(256)
void softmax16_kernel(float* __restrict__ attn) {
    const int t = blockIdx.x * blockDim.x + threadIdx.x;   // 0 .. NROW*NH-1
    if (t >= NROW * NH) return;
    float* p = attn + (size_t)t * 16;
    float v[16];
    float mx = -3.4e38f;
#pragma unroll
    for (int i = 0; i < 16; ++i) { v[i] = p[i]; mx = fmaxf(mx, v[i]); }
    float s = 0.f;
#pragma unroll
    for (int i = 0; i < 16; ++i) { v[i] = __expf(v[i] - mx); s += v[i]; }
    const float inv = 1.0f / s;
#pragma unroll
    for (int i = 0; i < 16; ++i) p[i] = v[i] * inv;
}

// ---------------------------------------------------------------------------
// Deformable sampling: one wave per (b, q, h); lane = channel d (HD == 32).
// v is projected-value in bf16, (bs, nv, C) head-major channels -> each corner
// gather is one coalesced 64B transaction, streaming from the 192MB L2.
// ---------------------------------------------------------------------------
__global__ __launch_bounds__(256)
void msda_sample_kernel(const __bf16* __restrict__ v,     // (BS*NV, C) bf16
                        const float* __restrict__ off,    // (NROW, 256)
                        const float* __restrict__ attn,   // (NROW, 128) softmaxed
                        const float* __restrict__ refp,   // (BS, NQ, NP, 2)
                        float* __restrict__ out) {        // (NROW, 256)
    const int lane = threadIdx.x & 31;                 // channel d
    const int gw   = blockIdx.x * 8 + (threadIdx.x >> 5);
    const int h    = gw & 7;
    const int row  = gw >> 3;                          // 0 .. NROW-1
    const int b    = row / NQ;
    const int q    = row - b * NQ;

    const int Hs[NL] = {128, 64, 32, 16};
    const int Ws[NL] = {128, 64, 32, 16};
    const int St[NL] = {0, 16384, 20480, 21504};

    const float* offr  = off  + (size_t)row * (NH * NL * NP * 2);
    const float* attnr = attn + (size_t)row * (NH * NL * NP);
    const float* refr  = refp + ((size_t)b * NQ + q) * (NP * 2);
    const __bf16* vb   = v + (size_t)b * NV * CDIM + h * HD + lane;

    float acc = 0.f;
#pragma unroll
    for (int l = 0; l < NL; ++l) {
        const int   Hl = Hs[l], Wl = Ws[l], st = St[l];
        const float fW = (float)Wl, fH = (float)Hl;
#pragma unroll
        for (int p = 0; p < NP; ++p) {
            const float rx = refr[p * 2 + 0];
            const float ry = refr[p * 2 + 1];
            const int   oc = ((h * NL + l) * NP + p) * 2;
            const float ox = offr[oc + 0];
            const float oy = offr[oc + 1];
            const float aw = attnr[h * (NL * NP) + l * NP + p];

            // loc = ref + off / (Wl, Hl);  pixel = loc * dim - 0.5
            const float x = (rx + ox / fW) * fW - 0.5f;
            const float y = (ry + oy / fH) * fH - 0.5f;

            const float x0f = floorf(x), y0f = floorf(y);
            const float fx = x - x0f,    fy = y - y0f;
            const int   ix0 = (int)x0f,  iy0 = (int)y0f;

            float s = 0.f;
#pragma unroll
            for (int cy = 0; cy < 2; ++cy) {
#pragma unroll
                for (int cx = 0; cx < 2; ++cx) {
                    const int ix = ix0 + cx, iy = iy0 + cy;
                    const float wgt = (cx ? fx : 1.f - fx) * (cy ? fy : 1.f - fy);
                    if (ix >= 0 && ix < Wl && iy >= 0 && iy < Hl) {
                        const size_t idx = (size_t)(st + iy * Wl + ix) * CDIM;
                        s += wgt * (float)vb[idx];
                    }
                }
            }
            acc += aw * s;
        }
    }
    out[(size_t)row * CDIM + h * HD + lane] = acc;   // head-major concat
}

// ---------------------------------------------------------------------------
// Host-side launcher
// ---------------------------------------------------------------------------
extern "C" void kernel_launch(void* const* d_in, const int* in_sizes, int n_in,
                              void* d_out, int out_size, void* d_ws, size_t ws_size,
                              hipStream_t stream) {
    const float* query  = (const float*)d_in[0];   // (8,2000,256)
    const float* value  = (const float*)d_in[1];   // (8,21760,256)
    const float* refp   = (const float*)d_in[2];   // (8,2000,4,2)
    const float* w_off  = (const float*)d_in[3];   // (256,256)
    const float* b_off  = (const float*)d_in[4];   // (256,)
    const float* w_attn = (const float*)d_in[5];   // (256,128)
    const float* b_attn = (const float*)d_in[6];   // (128,)
    const float* w_val  = (const float*)d_in[7];   // (256,256)
    const float* b_val  = (const float*)d_in[8];   // (256,)
    const float* w_out  = (const float*)d_in[9];   // (256,256)
    const float* b_out  = (const float*)d_in[10];  // (256,)
    // d_in[11] = spatial_shapes (static, hardcoded in kernels)

    // Workspace layout: fp32 sections, then bf16 value tensor, then packed weights.
    float* ws        = (float*)d_ws;
    float* off_buf   = ws;                                     // NROW*256 f32
    float* attn_buf  = off_buf  + (size_t)NROW * 256;          // NROW*128 f32
    float* samp_buf  = attn_buf + (size_t)NROW * 128;          // NROW*256 f32
    __bf16* v_bf16   = (__bf16*)(samp_buf + (size_t)NROW * 256);   // MV*256 bf16
    __bf16* pw_val   = v_bf16 + (size_t)MV * CDIM;             // 256*256
    __bf16* pw_off   = pw_val + (size_t)CDIM * CDIM;           // 256*256
    __bf16* pw_attn  = pw_off + (size_t)CDIM * CDIM;           // 256*128
    __bf16* pw_out   = pw_attn + (size_t)CDIM * 128;           // 256*256

    const dim3 blk(256);

    // 0) One-time weight pre-swizzle into WMMA B-fragment layout (bf16).
    pack_weight_bf16<<<(4096 + 255) / 256, blk, 0, stream>>>(w_val,  pw_val,  CDIM, CDIM);
    pack_weight_bf16<<<(4096 + 255) / 256, blk, 0, stream>>>(w_off,  pw_off,  CDIM, CDIM);
    pack_weight_bf16<<<(2048 + 255) / 256, blk, 0, stream>>>(w_attn, pw_attn, CDIM, 128);
    pack_weight_bf16<<<(4096 + 255) / 256, blk, 0, stream>>>(w_out,  pw_out,  CDIM, CDIM);

    // 1) Value projection: v = value @ w_val + b_val  (bf16 output: halves HBM bytes)
    gemm_bias_wmma<true, CDIM><<<dim3(CDIM / 64, MV / 64), blk, 0, stream>>>(
        value, pw_val, b_val, v_bf16, MV, CDIM);

    // 2) Sampling offsets: off = query @ w_off + b_off
    gemm_bias_wmma<false, CDIM><<<dim3(CDIM / 64, NROW / 64), blk, 0, stream>>>(
        query, pw_off, b_off, off_buf, NROW, CDIM);

    // 3) Attention logits: attn = query @ w_attn + b_attn  (N = 128)
    gemm_bias_wmma<false, CDIM><<<dim3(128 / 64, NROW / 64), blk, 0, stream>>>(
        query, pw_attn, b_attn, attn_buf, NROW, 128);

    // 4) Softmax over the 16 (level, point) logits per (row, head)
    softmax16_kernel<<<(NROW * NH + 255) / 256, blk, 0, stream>>>(attn_buf);

    // 5) Bilinear sampling + weighted accumulation; one wave per (b,q,h)
    msda_sample_kernel<<<NROW * NH / 8, blk, 0, stream>>>(
        v_bf16, off_buf, attn_buf, refp, samp_buf);

    // 6) Output projection: out = samp @ w_out + b_out
    gemm_bias_wmma<false, CDIM><<<dim3(CDIM / 64, NROW / 64), blk, 0, stream>>>(
        samp_buf, pw_out, b_out, (float*)d_out, NROW, CDIM);
}
